// CrossModal_Ranked_Attention_57062935494765
// MI455X (gfx1250) — matrix operand order repaired and checked by reading, not compiled, not run on hardware
//
#include <hip/hip_runtime.h>
#include <hip/hip_bf16.h>

// ---------------------------------------------------------------------------
// CrossModal ranked attention for MI455X (gfx1250, wave32, WMMA bf16).
// Heavy math -> v_wmma_f32_16x16x32_bf16 (f32 accumulate).
// Weight tiles staged in LDS with double-buffered GLOBAL_LOAD_ASYNC_TO_LDS_B128
// (ASYNCcnt / s_wait_asynccnt), shared by all 8 waves of the workgroup.
// DS->WMMA fragment reads are software-pipelined 2 deep.
// ---------------------------------------------------------------------------

typedef __bf16 bf16_t;
typedef __attribute__((ext_vector_type(16))) __bf16 v16bf;
typedef __attribute__((ext_vector_type(8)))  float  v8f;

#define BB    65536
#define DDT   768
#define DDIM  2048
#define DD    300
#define NPAD  304    // 19 * 16
#define KQK   320    // 300 padded up to multiple of 32
#define NT    19     // n-tiles of 16 covering 304

// LDS tile geometry (byte strides padded for conflict-free ds_load_b128)
#define PROJ_ROWB   80            // 32 bf16 (64B) + 16B pad
#define PROJ_TILE   (NPAD * PROJ_ROWB)        // 24320 B
#define SC_ROWB     656           // 320 bf16 (640B) + 16B pad
#define SC_MATB     (16 * SC_ROWB)            // 10496 B
#define SC_TILE     (2 * SC_MATB)             // Wq + Wk, 20992 B

union Frag {
  v16bf v;
  uint4 q[2];
};

static __device__ __forceinline__ v8f wmma_bf16(v16bf a, v16bf b, v8f c) {
  return __builtin_amdgcn_wmma_f32_16x16x32_bf16(false, a, false, b, (short)0, c,
                                                 false, false);
}

// Async copy 16 bytes global -> LDS (tracked by ASYNCcnt).
static __device__ __forceinline__ void async_b128(unsigned lds_off, const void* g) {
  asm volatile("global_load_async_to_lds_b128 %0, %1, off"
               :: "v"(lds_off), "v"(g) : "memory");
}
#define WAIT_ASYNC(n) asm volatile("s_wait_asynccnt %0" :: "n"(n) : "memory")

// ---------------------------------------------------------------------------
// Weight prep: f32 [K][300] -> bf16 [NPAD][Kpad] (transposed, zero padded).
// ---------------------------------------------------------------------------
__global__ void k_transpose_pad(const float* __restrict__ src, bf16_t* __restrict__ dst,
                                int Kpad, int Kvalid) {
  int idx = blockIdx.x * blockDim.x + threadIdx.x;
  int total = NPAD * Kpad;
  if (idx >= total) return;
  int n = idx / Kpad;
  int k = idx - n * Kpad;
  float v = (n < DD && k < Kvalid) ? src[k * DD + n] : 0.0f;
  dst[idx] = (bf16_t)v;
}

// ---------------------------------------------------------------------------
// proj = X[B,DIN] @ W[DIN,300] + b.
// Block = 8 waves, each wave owns a 16-row m-tile and all 19 n-tiles.
// Per k-step: async-stage W[0..303][kb..kb+32) (19.5KB, double buffered);
// B fragments read from LDS, pipelined 2 deep against the WMMA chain.
// ---------------------------------------------------------------------------
template <int DIN>
__global__ void __launch_bounds__(256)
k_proj(const float* __restrict__ X,
       const bf16_t* __restrict__ WT,   // [NPAD][DIN]
       const float* __restrict__ bias,  // [300]
       float* __restrict__ outF,        // [B][300] (d_out segment)
       bf16_t* __restrict__ outBF) {    // [B][KQK], cols 300..319 zeroed
  __shared__ __align__(16) unsigned char smem[2 * PROJ_TILE];
  const unsigned smem_base = (unsigned)(uintptr_t)&smem[0];

  const int tid  = threadIdx.x;
  const int lane = tid & 31;
  const int wave = tid >> 5;
  const int row0 = (blockIdx.x * 8 + wave) * 16;
  const int half = lane >> 4;
  const int lm   = lane & 15;

  const int STEPS = DIN / 32;

  // --- tile staging: 304 rows x 4 chunks of 16B = 1216 transfers, 5/thread
  auto stage = [&](int bufIdx, int kb) {
    const unsigned base = smem_base + (unsigned)bufIdx * PROJ_TILE;
#pragma unroll
    for (int i = 0; i < 5; ++i) {
      int e = tid + i * 256;
      int r = e >> 2;
      int c = e & 3;
      if (r > NPAD - 1) r = NPAD - 1;   // clamp: duplicate write, same data
      async_b128(base + (unsigned)(r * PROJ_ROWB + c * 16),
                 WT + (size_t)r * DIN + kb + c * 8);
    }
  };

  const v8f vzero = {};
  v8f acc[NT];
#pragma unroll
  for (int t = 0; t < NT; ++t) acc[t] = vzero;

  const float* xrow = X + (size_t)(row0 + lm) * DIN;
  const unsigned fragBase = (unsigned)(lm * PROJ_ROWB + half * 32);

  stage(0, 0);

#pragma unroll 1
  for (int ks = 0; ks < STEPS; ++ks) {
    const int kb = ks * 32;

    // A fragment (contiguous k order, same as the B reads): lane<16 holds K {kb..kb+15}, lane>=16 K {kb+16..kb+31}
    Frag a;
    const int k0 = kb + half * 16;
#pragma unroll
    for (int c = 0; c < 2; ++c) {
      const float4 f0 = *(const float4*)(xrow + k0 + c * 8);
      const float4 f1 = *(const float4*)(xrow + k0 + c * 8 + 4);
      a.v[8 * c + 0] = (bf16_t)f0.x; a.v[8 * c + 1] = (bf16_t)f0.y;
      a.v[8 * c + 2] = (bf16_t)f0.z; a.v[8 * c + 3] = (bf16_t)f0.w;
      a.v[8 * c + 4] = (bf16_t)f1.x; a.v[8 * c + 5] = (bf16_t)f1.y;
      a.v[8 * c + 6] = (bf16_t)f1.z; a.v[8 * c + 7] = (bf16_t)f1.w;
    }

    if (ks + 1 < STEPS) {
      stage((ks + 1) & 1, kb + 32);   // prefetch next tile (async)
      WAIT_ASYNC(5);                  // our 5 loads for the current tile done
    } else {
      WAIT_ASYNC(0);
    }
    __syncthreads();                  // all waves staged current tile

    const unsigned cb = fragBase + (unsigned)(ks & 1) * PROJ_TILE;

    // 2-deep pipelined fragment reads: load b[t+1] while WMMA consumes b[t].
    Frag b[2];
    b[0].q[0] = *(const uint4*)&smem[cb];
    b[0].q[1] = *(const uint4*)&smem[cb + 16];
#pragma unroll
    for (int t = 0; t < NT; ++t) {
      const int cur = t & 1;
      if (t + 1 < NT) {
        const unsigned nb = cb + (unsigned)((t + 1) * 16 * PROJ_ROWB);
        b[cur ^ 1].q[0] = *(const uint4*)&smem[nb];
        b[cur ^ 1].q[1] = *(const uint4*)&smem[nb + 16];
      }
      acc[t] = wmma_bf16(a.v, b[cur].v, acc[t]);
    }
    __syncthreads();                  // tile consumed; safe to overwrite
  }

  // Epilogue: C layout -> VGPR v is row (v + half*8), lane column = n.
#pragma unroll
  for (int t = 0; t < NT; ++t) {
    const int n = t * 16 + lm;
    const float bv = (n < DD) ? bias[n] : 0.0f;
#pragma unroll
    for (int v = 0; v < 8; ++v) {
      const int row = row0 + half * 8 + v;
      const float val = acc[t][v] + bv;
      if (n < DD) outF[(size_t)row * DD + n] = val;
      outBF[(size_t)row * KQK + n] = (bf16_t)val;
    }
  }
  // Zero the K-pad region [304,320) so the score GEMM pad is clean.
  {
    const uint4 z = {0u, 0u, 0u, 0u};
    *(uint4*)(outBF + (size_t)(row0 + lm) * KQK + NPAD + half * 8) = z;
  }
}

// ---------------------------------------------------------------------------
// Z = sigmoid( (P@Wq+bq) . (P@Wk+bk) / sqrt(300) ) per row.
// A fragments (16 rows x K=320) preloaded in registers; per n-tile the Wq/Wk
// tiles (20.5KB) are async-staged into LDS (double buffered); the 20 WMMAs
// per tile consume 2-deep pipelined LDS fragment reads.
// ---------------------------------------------------------------------------
__global__ void __launch_bounds__(256)
k_score(const bf16_t* __restrict__ P,    // [B][KQK]
        const bf16_t* __restrict__ Wq,   // [NPAD][KQK] transposed
        const bf16_t* __restrict__ Wk,   // [NPAD][KQK] transposed
        const float* __restrict__ bq,
        const float* __restrict__ bk,
        float* __restrict__ Z) {         // [B]
  __shared__ __align__(16) unsigned char smem[2 * SC_TILE];
  const unsigned smem_base = (unsigned)(uintptr_t)&smem[0];

  const int tid  = threadIdx.x;
  const int lane = tid & 31;
  const int wave = tid >> 5;
  const int row0 = (blockIdx.x * 8 + wave) * 16;
  const int half = lane >> 4;
  const int lm   = lane & 15;

  // --- stage Wq+Wk tiles for n-tile nt: 2 mats x 16 rows x 40 chunks = 1280
  auto stage = [&](int bufIdx, int nt) {
    const unsigned base = smem_base + (unsigned)bufIdx * SC_TILE;
#pragma unroll
    for (int i = 0; i < 5; ++i) {
      int e = tid + i * 256;
      int mat = e >= 640;
      int rem = e - mat * 640;
      int r = rem / 40;
      int c = rem - r * 40;
      const bf16_t* src = (mat ? Wk : Wq) + (size_t)(nt * 16 + r) * KQK + c * 8;
      async_b128(base + (unsigned)(mat * SC_MATB + r * SC_ROWB + c * 16), src);
    }
  };

  // Preload this wave's A fragments for all 10 k-steps (K = 320).
  const bf16_t* prow = P + (size_t)(row0 + lm) * KQK;
  Frag a[KQK / 32];
#pragma unroll
  for (int ks = 0; ks < KQK / 32; ++ks) {
    const int k0 = ks * 32 + half * 16;
    a[ks].q[0] = *(const uint4*)(prow + k0);
    a[ks].q[1] = *(const uint4*)(prow + k0 + 8);
  }

  const v8f vzero = {};
  float s[8];
#pragma unroll
  for (int v = 0; v < 8; ++v) s[v] = 0.0f;

  const unsigned fragBase = (unsigned)(lm * SC_ROWB + half * 32);

  stage(0, 0);

#pragma unroll 1
  for (int t = 0; t < NT; ++t) {
    if (t + 1 < NT) {
      stage((t + 1) & 1, t + 1);
      WAIT_ASYNC(5);
    } else {
      WAIT_ASYNC(0);
    }
    __syncthreads();

    const unsigned cb = fragBase + (unsigned)((t & 1) * SC_TILE);
    v8f qa = vzero, ka = vzero;

    // 2-deep pipelined fragment reads.
    Frag fq[2], fk[2];
    fq[0].q[0] = *(const uint4*)&smem[cb];
    fq[0].q[1] = *(const uint4*)&smem[cb + 16];
    fk[0].q[0] = *(const uint4*)&smem[cb + SC_MATB];
    fk[0].q[1] = *(const uint4*)&smem[cb + SC_MATB + 16];
#pragma unroll
    for (int ks = 0; ks < KQK / 32; ++ks) {
      const int cur = ks & 1;
      if (ks + 1 < KQK / 32) {
        const unsigned ro = cb + (unsigned)((ks + 1) * 64);
        fq[cur ^ 1].q[0] = *(const uint4*)&smem[ro];
        fq[cur ^ 1].q[1] = *(const uint4*)&smem[ro + 16];
        fk[cur ^ 1].q[0] = *(const uint4*)&smem[ro + SC_MATB];
        fk[cur ^ 1].q[1] = *(const uint4*)&smem[ro + SC_MATB + 16];
      }
      qa = wmma_bf16(a[ks].v, fq[cur].v, qa);
      ka = wmma_bf16(a[ks].v, fk[cur].v, ka);
    }

    const int n = t * 16 + lm;
    const float bqv = (n < DD) ? bq[n] : 0.0f;
    const float bkv = (n < DD) ? bk[n] : 0.0f;
#pragma unroll
    for (int v = 0; v < 8; ++v)
      s[v] += (qa[v] + bqv) * (ka[v] + bkv);

    __syncthreads();  // tile consumed; next iteration may overwrite
  }

  // Reduce over the 16 lanes of each half (lane = column index).
#pragma unroll
  for (int v = 0; v < 8; ++v) {
#pragma unroll
    for (int m = 8; m >= 1; m >>= 1)
      s[v] += __shfl_xor(s[v], m, 16);
  }

  if (lm == 0) {
    const float inv_sqrt_d = 0.05773502691896258f;  // 1/sqrt(300)
#pragma unroll
    for (int v = 0; v < 8; ++v) {
      const float sc = s[v] * inv_sqrt_d;
      Z[row0 + half * 8 + v] = 1.0f / (1.0f + __expf(-sc));
    }
  }
}

// ---------------------------------------------------------------------------
// softmax([ZI*ZT, ZCD*ZT]) over 2 == sigmoid of the logit difference.
// ---------------------------------------------------------------------------
__global__ void __launch_bounds__(320)
k_scale(const float* __restrict__ ZT, const float* __restrict__ ZI,
        const float* __restrict__ ZCD,
        float* __restrict__ outIM, float* __restrict__ outCD) {
  const int row = blockIdx.x;
  const int n = threadIdx.x;
  if (n >= DD) return;
  const float zt = ZT[row];
  const float a = ZI[row] * zt;
  const float b = ZCD[row] * zt;
  const float w1 = 1.0f / (1.0f + __expf(b - a));
  const float w2 = 1.0f - w1;
  const size_t i = (size_t)row * DD + n;
  outIM[i] *= w1;
  outCD[i] *= w2;
}

// ---------------------------------------------------------------------------
extern "C" void kernel_launch(void* const* d_in, const int* in_sizes, int n_in,
                              void* d_out, int out_size, void* d_ws, size_t ws_size,
                              hipStream_t stream) {
  const float* T   = (const float*)d_in[0];
  const float* IM  = (const float*)d_in[1];
  const float* CD  = (const float*)d_in[2];
  const float* Wt  = (const float*)d_in[3];
  const float* bt  = (const float*)d_in[4];
  const float* Wim = (const float*)d_in[5];
  const float* bim = (const float*)d_in[6];
  const float* WqT_f = (const float*)d_in[7];
  const float* bqT   = (const float*)d_in[8];
  const float* WkT_f = (const float*)d_in[9];
  const float* bkT   = (const float*)d_in[10];
  const float* WqI_f = (const float*)d_in[11];
  const float* bqI   = (const float*)d_in[12];
  const float* WkI_f = (const float*)d_in[13];
  const float* bkI   = (const float*)d_in[14];
  const float* WqCD_f = (const float*)d_in[15];
  const float* bqCD   = (const float*)d_in[16];
  const float* WkCD_f = (const float*)d_in[17];
  const float* bkCD   = (const float*)d_in[18];

  float* out   = (float*)d_out;
  float* outT  = out;
  float* outIM = out + (size_t)BB * DD;
  float* outCD = out + 2 * (size_t)BB * DD;

  char* ws = (char*)d_ws;
  size_t off = 0;
  auto alloc = [&](size_t bytes) -> void* {
    void* p = ws + off;
    off = (off + bytes + 255) & ~(size_t)255;
    return p;
  };

  bf16_t* wtT    = (bf16_t*)alloc((size_t)NPAD * DDT * 2);
  bf16_t* wimT   = (bf16_t*)alloc((size_t)NPAD * DDIM * 2);
  bf16_t* wqT_t  = (bf16_t*)alloc((size_t)NPAD * KQK * 2);
  bf16_t* wkT_t  = (bf16_t*)alloc((size_t)NPAD * KQK * 2);
  bf16_t* wqI_t  = (bf16_t*)alloc((size_t)NPAD * KQK * 2);
  bf16_t* wkI_t  = (bf16_t*)alloc((size_t)NPAD * KQK * 2);
  bf16_t* wqCD_t = (bf16_t*)alloc((size_t)NPAD * KQK * 2);
  bf16_t* wkCD_t = (bf16_t*)alloc((size_t)NPAD * KQK * 2);
  bf16_t* pT  = (bf16_t*)alloc((size_t)BB * KQK * 2);
  bf16_t* pIM = (bf16_t*)alloc((size_t)BB * KQK * 2);
  bf16_t* pCD = (bf16_t*)alloc((size_t)BB * KQK * 2);
  float* ZT  = (float*)alloc((size_t)BB * 4);
  float* ZI  = (float*)alloc((size_t)BB * 4);
  float* ZCD = (float*)alloc((size_t)BB * 4);

  // 1) Weight prep (transpose + bf16 + pad). All tiny.
  k_transpose_pad<<<(NPAD * DDT + 255) / 256, 256, 0, stream>>>(Wt, wtT, DDT, DDT);
  k_transpose_pad<<<(NPAD * DDIM + 255) / 256, 256, 0, stream>>>(Wim, wimT, DDIM, DDIM);
  const int qkBlocks = (NPAD * KQK + 255) / 256;
  k_transpose_pad<<<qkBlocks, 256, 0, stream>>>(WqT_f,  wqT_t,  KQK, DD);
  k_transpose_pad<<<qkBlocks, 256, 0, stream>>>(WkT_f,  wkT_t,  KQK, DD);
  k_transpose_pad<<<qkBlocks, 256, 0, stream>>>(WqI_f,  wqI_t,  KQK, DD);
  k_transpose_pad<<<qkBlocks, 256, 0, stream>>>(WkI_f,  wkI_t,  KQK, DD);
  k_transpose_pad<<<qkBlocks, 256, 0, stream>>>(WqCD_f, wqCD_t, KQK, DD);
  k_transpose_pad<<<qkBlocks, 256, 0, stream>>>(WkCD_f, wkCD_t, KQK, DD);

  // 2) Projections (WMMA, async-LDS weight staging).
  k_proj<DDT><<<512, 256, 0, stream>>>(T,  wtT,  bt,  outT,  pT);
  k_proj<DDIM><<<512, 256, 0, stream>>>(IM, wimT, bim, outIM, pIM);
  k_proj<DDT><<<512, 256, 0, stream>>>(CD, wtT,  bt,  outCD, pCD);

  // 3) Branch scores (WMMA q/k + per-row dot + sigmoid).
  k_score<<<512, 256, 0, stream>>>(pT,  wqT_t,  wkT_t,  bqT,  bkT,  ZT);
  k_score<<<512, 256, 0, stream>>>(pIM, wqI_t,  wkI_t,  bqI,  bkI,  ZI);
  k_score<<<512, 256, 0, stream>>>(pCD, wqCD_t, wkCD_t, bqCD, bkCD, ZCD);

  // 4) 2-way softmax + in-place scaling of d_out segments 1 and 2.
  k_scale<<<BB, 320, 0, stream>>>(ZT, ZI, ZCD, outIM, outCD);
}
